// EnhancedSetAbstraction_8881992368320
// MI455X (gfx1250) — matrix operand order repaired and checked by reading, not compiled, not run on hardware
//
#include <hip/hip_runtime.h>
#include <hip/hip_bf16.h>
#include <math.h>

typedef __attribute__((ext_vector_type(16))) __bf16 bf16x16;
typedef __attribute__((ext_vector_type(8)))  float  f32x8;

// ================= generic batched bf16 WMMA GEMM, 64x64 macro-tile =================
// C[M,N] = act((A[M,K] @ B[K,N] + bias[N]) * scale[N] + shift[N])
// A,B bf16 row-major (lda/ldb); C f32 or bf16 (ldc). Batched over
// grid.z (outer, strides s?b) and grid.y (inner "head", strides s?h).
// M,N multiples of 64; K multiple of 32 (guaranteed by caller).
// Each wave owns a 64x64 tile = 4x4 WMMA sub-tiles: 16 v_wmma per 32-deep
// K-step against 8 b128 A loads and 64 strided 16-bit B loads.
__global__ __launch_bounds__(256)
void gemm_bf16_wmma(const __bf16* __restrict__ A, const __bf16* __restrict__ Bm,
                    void* __restrict__ Cp,
                    const float* __restrict__ bias, const float* __restrict__ scale,
                    const float* __restrict__ shift,
                    int M, int N, int K, int lda, int ldb, int ldc,
                    long long sAh, long long sBh, long long sCh,
                    long long sAb, long long sBb, long long sCb,
                    int relu, int out_bf16)
{
  const int h = blockIdx.y, bb = blockIdx.z;
  const __bf16* Ab = A  + (long long)bb * sAb + (long long)h * sAh;
  const __bf16* Bb = Bm + (long long)bb * sBb + (long long)h * sBh;
  const long long cOff = (long long)bb * sCb + (long long)h * sCh;

  const int lane = threadIdx.x & 31;
  const int wave = threadIdx.x >> 5;
  const int tilesN = N >> 6;                       // 64-wide macro tiles
  const int tile = blockIdx.x * 8 + wave;
  if (tile >= (M >> 6) * tilesN) return;           // whole-wave uniform exit
  const int tm = (tile / tilesN) << 6;
  const int tn = (tile % tilesN) << 6;

  const int m    = lane & 15;
  const int half = lane >> 4;
  // A fragment: lane holds row m; VGPR0-3 = k {0..7}+8*half, VGPR4-7 = k {16..23}+8*half
  const __bf16* arow = Ab + (long long)(tm + m) * lda + half * 8;
  // B fragment: lane holds column n = m; elements = k {0..7}+8*half, then k {16..23}+8*half
  const __bf16* brow = Bb + (long long)(half * 8) * ldb + tn + m;

  f32x8 acc[4][4] = {};
  for (int kb = 0; kb < K; kb += 32) {
    union { bf16x16 v; unsigned u[8]; } a[4];
    #pragma unroll
    for (int i = 0; i < 4; ++i) {
      const __bf16* ar = arow + (long long)(i * 16) * lda + kb;
      const unsigned* ap  = (const unsigned*)ar;
      a[i].u[0]=ap[0];  a[i].u[1]=ap[1];  a[i].u[2]=ap[2];  a[i].u[3]=ap[3];
      const unsigned* ap2 = (const unsigned*)(ar + 16);
      a[i].u[4]=ap2[0]; a[i].u[5]=ap2[1]; a[i].u[6]=ap2[2]; a[i].u[7]=ap2[3];
    }
    union { bf16x16 v; uint4 q[2]; } bq[4];
    const __bf16* br = brow + (long long)kb * ldb;
    #pragma unroll
    for (int j = 0; j < 4; ++j) {
      #pragma unroll
      for (int s = 0; s < 16; ++s) bq[j].v[s] = br[(long long)(s + (s & 8)) * ldb + j * 16];
    }
    __builtin_prefetch(arow + kb + 32, 0, 1);                  // global_prefetch_b8
    __builtin_prefetch(br + (long long)32 * ldb, 0, 1);
    #pragma unroll
    for (int i = 0; i < 4; ++i) {
      #pragma unroll
      for (int j = 0; j < 4; ++j) {
        acc[i][j] = __builtin_amdgcn_wmma_f32_16x16x32_bf16(false, a[i].v, false, bq[j].v,
                                                            (short)0, acc[i][j], false, false);
      }
    }
  }

  // C/D layout: lane -> N = lane&15, VGPR e -> M = e + 8*half
  const int nl = lane & 15;
  const int mbase = half << 3;
  #pragma unroll
  for (int j = 0; j < 4; ++j) {
    const int n = tn + j * 16 + nl;
    const float bi = bias  ? bias[n]  : 0.f;
    const float sc = scale ? scale[n] : 1.f;
    const float sh = shift ? shift[n] : 0.f;
    #pragma unroll
    for (int i = 0; i < 4; ++i) {
      const int mrow0 = tm + i * 16 + mbase;
      #pragma unroll
      for (int e = 0; e < 8; ++e) {
        float x = (acc[i][j][e] + bi) * sc + sh;
        if (relu) x = fmaxf(x, 0.f);
        const long long off2 = cOff + (long long)(mrow0 + e) * ldc + n;
        if (out_bf16) ((__bf16*)Cp)[off2] = (__bf16)x;
        else          ((float*)Cp)[off2]  = x;
      }
    }
  }
}

// ================= weight prep kernels =================
__global__ __launch_bounds__(256)
void k_cast_bf16(const float* __restrict__ s, __bf16* __restrict__ d, int n) {
  int i = blockIdx.x * 256 + threadIdx.x;
  if (i < n) d[i] = (__bf16)s[i];
}
// (K,N) f32 -> (Kp,N) bf16, zero padded rows
__global__ __launch_bounds__(256)
void k_pad_bf16(const float* __restrict__ s, __bf16* __restrict__ d, int K, int Kp, int N) {
  int i = blockIdx.x * 256 + threadIdx.x;
  if (i >= Kp * N) return;
  int k = i / N, n = i % N;
  d[i] = (k < K) ? (__bf16)s[k * N + n] : (__bf16)0.f;
}
// (O,C) f32 -> (Cp,O) bf16 transposed, zero padded rows
__global__ __launch_bounds__(256)
void k_tpad_bf16(const float* __restrict__ s, __bf16* __restrict__ d, int O, int C, int Cp) {
  int i = blockIdx.x * 256 + threadIdx.x;
  if (i >= Cp * O) return;
  int c = i / O, o = i % O;
  d[i] = (c < C) ? (__bf16)s[o * C + c] : (__bf16)0.f;
}
__global__ __launch_bounds__(256)
void k_bnaff(const float* __restrict__ g, const float* __restrict__ b,
             float* __restrict__ sc, float* __restrict__ sh, int n, float inv) {
  int i = blockIdx.x * 256 + threadIdx.x;
  if (i < n) { sc[i] = g[i] * inv; sh[i] = b[i]; }
}

// ================= time embedding + comb (B*N x 96 bf16) =================
__global__ __launch_bounds__(256)
void k_comb(const float* __restrict__ ts, const float* __restrict__ points,
            const float* __restrict__ te_w1, const float* __restrict__ te_b1,
            const float* __restrict__ te_w2, const float* __restrict__ te_b2,
            __bf16* __restrict__ comb)
{
  int idx = blockIdx.x * 256 + threadIdx.x;
  if (idx >= 8 * 8192) return;
  int b = idx >> 13, n = idx & 8191;
  float t = ts[idx];
  float tf[8];
  #pragma unroll
  for (int o = 0; o < 8; ++o) tf[o] = te_b2[o];
  #pragma unroll
  for (int j = 0; j < 4; ++j) {
    float hj = fmaxf(t * te_w1[j] + te_b1[j], 0.f);
    #pragma unroll
    for (int o = 0; o < 8; ++o) tf[o] += hj * te_w2[j * 8 + o];
  }
  #pragma unroll
  for (int d = 0; d < 4; ++d) {
    float ang = t * __expf(-(float)(2 * d) * 1.1512925465f);  // ln(1e4)/8
    tf[2 * d]     += __sinf(ang);
    tf[2 * d + 1] += __cosf(ang);
  }
  __bf16* row = comb + (long long)idx * 96;
  for (int c = 0; c < 64; ++c) row[c] = (__bf16)points[((long long)(b * 64 + c)) * 8192 + n];
  #pragma unroll
  for (int o = 0; o < 8; ++o) row[64 + o] = (__bf16)tf[o];
  #pragma unroll
  for (int c = 72; c < 96; ++c) row[c] = (__bf16)0.f;
}

// ================= sampling: softmax stats, gumbel keys, bitonic top-k =================
__device__ __forceinline__ unsigned hash_u32(unsigned x) {
  x ^= x >> 16; x *= 0x7feb352dU; x ^= x >> 15; x *= 0x846ca68bU; x ^= x >> 16; return x;
}

__global__ __launch_bounds__(256)
void k_stats(const float* __restrict__ ts, float* __restrict__ stats) {
  int b = blockIdx.x;
  __shared__ float red[256];
  float mx = -3.4e38f;
  for (int i = threadIdx.x; i < 8192; i += 256) mx = fmaxf(mx, ts[b * 8192 + i] * 10.f);
  red[threadIdx.x] = mx; __syncthreads();
  for (int o = 128; o > 0; o >>= 1) { if (threadIdx.x < o) red[threadIdx.x] = fmaxf(red[threadIdx.x], red[threadIdx.x + o]); __syncthreads(); }
  mx = red[0]; __syncthreads();
  float sm = 0.f;
  for (int i = threadIdx.x; i < 8192; i += 256) sm += __expf(ts[b * 8192 + i] * 10.f - mx);
  red[threadIdx.x] = sm; __syncthreads();
  for (int o = 128; o > 0; o >>= 1) { if (threadIdx.x < o) red[threadIdx.x] += red[threadIdx.x + o]; __syncthreads(); }
  if (threadIdx.x == 0) { stats[b * 2] = mx; stats[b * 2 + 1] = red[0]; }
}

__global__ __launch_bounds__(512)
void k_sample(const float* __restrict__ ts, const float* __restrict__ xyz,
              const float* __restrict__ stats, int* __restrict__ sampIdx,
              float* __restrict__ outXyz)
{
  __shared__ float key[8192];
  __shared__ unsigned short sid[8192];
  int b = blockIdx.x, tid = threadIdx.x;
  float m = stats[b * 2], s = stats[b * 2 + 1];
  for (int n = tid; n < 8192; n += 512) {
    float tv = ts[b * 8192 + n] * 10.f;
    float w  = __expf(tv - m) / s + 0.1f;
    unsigned hsh = hash_u32(((unsigned)(b * 8192 + n)) ^ 0x9E3779B9u);
    float u = fmaxf((float)(hsh >> 8) * (1.f / 16777216.f), 1e-10f);
    float g = -__logf(-__logf(u));
    key[n] = __logf(w) + g;
    sid[n] = (unsigned short)n;
  }
  __syncthreads();
  for (int k = 2; k <= 8192; k <<= 1) {
    for (int j = k >> 1; j > 0; j >>= 1) {
      for (int p = tid; p < 4096; p += 512) {
        int i   = ((p / j) * (j << 1)) + (p % j);
        int ixj = i + j;
        bool desc = ((i & k) == 0);
        float a = key[i], c = key[ixj];
        bool sw = desc ? (a < c) : (a > c);
        if (sw) {
          key[i] = c; key[ixj] = a;
          unsigned short t2 = sid[i]; sid[i] = sid[ixj]; sid[ixj] = t2;
        }
      }
      __syncthreads();
    }
  }
  for (int n = tid; n < 2048; n += 512) {
    int id = sid[n];
    sampIdx[b * 2048 + n] = id;
    outXyz[(b * 3 + 0) * 2048 + n] = xyz[(b * 3 + 0) * 8192 + id];
    outXyz[(b * 3 + 1) * 2048 + n] = xyz[(b * 3 + 1) * 8192 + id];
    outXyz[(b * 3 + 2) * 2048 + n] = xyz[(b * 3 + 2) * 8192 + id];
  }
}

// ================= fused KNN (no d2 matrix in HBM) =================
__global__ __launch_bounds__(256)
void k_knn(const float* __restrict__ xyz, const float* __restrict__ newXyz,
           int* __restrict__ nidx)
{
  __shared__ float sx[2048], sy[2048], sz[2048];
  int b = blockIdx.y;
  int q = blockIdx.x * 256 + threadIdx.x;       // 0..2047
  float qx = newXyz[(b * 3 + 0) * 2048 + q];
  float qy = newXyz[(b * 3 + 1) * 2048 + q];
  float qz = newXyz[(b * 3 + 2) * 2048 + q];
  float nd[32]; int ni[32];
  #pragma unroll
  for (int i = 0; i < 32; ++i) { nd[i] = 3.4e38f; ni[i] = 0; }
  float worst = 3.4e38f; int wpos = 0;
  for (int c0 = 0; c0 < 8192; c0 += 2048) {
    for (int i = threadIdx.x; i < 2048; i += 256) {
      sx[i] = xyz[(b * 3 + 0) * 8192 + c0 + i];
      sy[i] = xyz[(b * 3 + 1) * 8192 + c0 + i];
      sz[i] = xyz[(b * 3 + 2) * 8192 + c0 + i];
    }
    __syncthreads();
    for (int i = 0; i < 2048; ++i) {
      float dx = sx[i] - qx, dy = sy[i] - qy, dz = sz[i] - qz;
      float d = dx * dx + dy * dy + dz * dz;
      if (d < worst) {
        nd[wpos] = d; ni[wpos] = c0 + i;
        float w = -1.f; int wp = 0;
        #pragma unroll
        for (int t = 0; t < 32; ++t) if (nd[t] > w) { w = nd[t]; wp = t; }
        worst = w; wpos = wp;
      }
    }
    __syncthreads();
  }
  long long base = ((long long)(b * 2048 + q)) * 32;
  #pragma unroll
  for (int i = 0; i < 32; ++i) nidx[base + i] = ni[i];
}

// ================= gather -> conv input rows (chunk of 4096 points) =================
__global__ __launch_bounds__(256)
void k_gather(const float* __restrict__ xyz, const float* __restrict__ newXyz,
              const __bf16* __restrict__ ptsB, const int* __restrict__ nidx,
              __bf16* __restrict__ Xc, int p0)
{
  int idx = blockIdx.x * 256 + threadIdx.x;   // pl*32 + s
  if (idx >= 4096 * 32) return;
  int s = idx & 31, pl = idx >> 5;
  int p = p0 + pl;
  int b = p >> 11, n = p & 2047;
  int id = nidx[((long long)p) * 32 + s];
  float qx = newXyz[(b * 3 + 0) * 2048 + n];
  float qy = newXyz[(b * 3 + 1) * 2048 + n];
  float qz = newXyz[(b * 3 + 2) * 2048 + n];
  __bf16* row = Xc + (long long)idx * 96;
  row[0] = (__bf16)(xyz[(b * 3 + 0) * 8192 + id] - qx);
  row[1] = (__bf16)(xyz[(b * 3 + 1) * 8192 + id] - qy);
  row[2] = (__bf16)(xyz[(b * 3 + 2) * 8192 + id] - qz);
  const __bf16* ps = ptsB + ((long long)(b * 8192 + id)) * 64;
  for (int c = 0; c < 64; ++c) row[3 + c] = ps[c];
  #pragma unroll
  for (int c = 67; c < 96; ++c) row[c] = (__bf16)0.f;
}

__global__ __launch_bounds__(256)
void k_maxpool(const __bf16* __restrict__ H3, float* __restrict__ feat, int p0)
{
  int idx = blockIdx.x * 256 + threadIdx.x;   // pl*256 + c
  if (idx >= 4096 * 256) return;
  int c = idx & 255, pl = idx >> 8;
  const __bf16* hh = H3 + (long long)pl * 32 * 256 + c;
  float m = -3.4e38f;
  for (int s = 0; s < 32; ++s) m = fmaxf(m, (float)hh[s * 256]);
  feat[((long long)(p0 + pl)) * 256 + c] = m;
}

// ================= attention helpers =================
__global__ __launch_bounds__(256)
void k_pool(const float* __restrict__ feat, __bf16* __restrict__ wx, int nw, int sc)
{
  int idx = blockIdx.x * 256 + threadIdx.x;   // b*nw*256
  if (idx >= 8 * nw * 256) return;
  int c = idx & 255;
  int j = (idx >> 8) % nw;
  int b = (idx >> 8) / nw;
  float sum = 0.f;
  for (int s = 0; s < sc; ++s) sum += feat[((long long)(b * 2048 + j * sc + s)) * 256 + c];
  wx[idx] = (__bf16)(sum / (float)sc);
}

__global__ __launch_bounds__(256)
void k_transK(const __bf16* __restrict__ Kb, __bf16* __restrict__ Kt, int nw)
{
  int idx = blockIdx.x * 256 + threadIdx.x;   // ((b*4+h)*64+dk)*nw + t
  if (idx >= 8 * 4 * 64 * nw) return;
  int t = idx % nw;
  int r = idx / nw;
  int dk = r & 63;
  int h  = (r >> 6) & 3;
  int b  = r >> 8;
  Kt[idx] = Kb[((long long)(b * nw + t)) * 256 + h * 64 + dk];
}

__global__ __launch_bounds__(256)
void k_softmax(const float* __restrict__ S, __bf16* __restrict__ P, int nw)
{
  int row = blockIdx.x;
  const float* sr = S + (long long)row * nw;
  __shared__ float red[256];
  float mx = -3.4e38f;
  for (int i = threadIdx.x; i < nw; i += 256) mx = fmaxf(mx, sr[i] * 0.125f);
  red[threadIdx.x] = mx; __syncthreads();
  for (int o = 128; o > 0; o >>= 1) { if (threadIdx.x < o) red[threadIdx.x] = fmaxf(red[threadIdx.x], red[threadIdx.x + o]); __syncthreads(); }
  mx = red[0]; __syncthreads();
  float sm = 0.f;
  for (int i = threadIdx.x; i < nw; i += 256) sm += __expf(sr[i] * 0.125f - mx);
  red[threadIdx.x] = sm; __syncthreads();
  for (int o = 128; o > 0; o >>= 1) { if (threadIdx.x < o) red[threadIdx.x] += red[threadIdx.x + o]; __syncthreads(); }
  float inv = 1.f / red[0];
  for (int i = threadIdx.x; i < nw; i += 256)
    P[(long long)row * nw + i] = (__bf16)(__expf(sr[i] * 0.125f - mx) * inv);
}

__global__ __launch_bounds__(256)
void k_interp(const float* __restrict__ attno, const float* __restrict__ swp,
              float* __restrict__ ms, int nw, int si, int first)
{
  int idx = blockIdx.x * 256 + threadIdx.x;   // b*2048*256
  if (idx >= 8 * 2048 * 256) return;
  int c = idx & 255;
  int t = (idx >> 8) & 2047;
  int b = idx >> 19;
  float pos = ((float)t + 0.5f) * ((float)nw / 2048.f) - 0.5f;
  pos = fminf(fmaxf(pos, 0.f), (float)(nw - 1));
  int lo = (int)floorf(pos);
  int hi = lo + 1 < nw ? lo + 1 : nw - 1;
  float fr = pos - (float)lo;
  float val = attno[((long long)(b * nw + lo)) * 256 + c] * (1.f - fr)
            + attno[((long long)(b * nw + hi)) * 256 + c] * fr;
  float sw = swp[si];
  if (first) ms[idx] = val * sw;
  else       ms[idx] += val * sw;
}

// ================= residual + layernorm, write transposed output =================
__global__ __launch_bounds__(256)
void k_layernorm(const float* __restrict__ feat, const float* __restrict__ ms,
                 const float* __restrict__ g, const float* __restrict__ bta,
                 float* __restrict__ outp)
{
  int tok = blockIdx.x;        // b*2048 + t
  int b = tok >> 11, t = tok & 2047;
  int c = threadIdx.x;
  float v = feat[(long long)tok * 256 + c] + ms[(long long)tok * 256 + c] * (1.f / 3.f);
  __shared__ float red[256];
  red[c] = v; __syncthreads();
  for (int o = 128; o > 0; o >>= 1) { if (c < o) red[c] += red[c + o]; __syncthreads(); }
  float mu = red[0] * (1.f / 256.f); __syncthreads();
  float d = v - mu;
  red[c] = d * d; __syncthreads();
  for (int o = 128; o > 0; o >>= 1) { if (c < o) red[c] += red[c + o]; __syncthreads(); }
  float var = red[0] * (1.f / 256.f);
  float y = d * rsqrtf(var + 1e-5f) * g[c] + bta[c];
  outp[((long long)(b * 256 + c)) * 2048 + t] = y;
}

// ================= host orchestration =================
static inline void gemm(hipStream_t st, const __bf16* A, const __bf16* B, void* C,
                        const float* bias, const float* scale, const float* shift,
                        int M, int N, int K, int lda, int ldb, int ldc,
                        int relu, int obf,
                        int nH = 1, long long sAh = 0, long long sBh = 0, long long sCh = 0,
                        int nB = 1, long long sAb = 0, long long sBb = 0, long long sCb = 0)
{
  int tiles = (M / 64) * (N / 64);                 // 64x64 macro-tiles, 8 waves/block
  dim3 grid((tiles + 7) / 8, nH, nB), block(256);
  gemm_bf16_wmma<<<grid, block, 0, st>>>(A, B, C, bias, scale, shift,
                                         M, N, K, lda, ldb, ldc,
                                         sAh, sBh, sCh, sAb, sBb, sCb, relu, obf);
}

extern "C" void kernel_launch(void* const* d_in, const int* in_sizes, int n_in,
                              void* d_out, int out_size, void* d_ws, size_t ws_size,
                              hipStream_t stream)
{
  (void)in_sizes; (void)n_in; (void)out_size; (void)ws_size;
  const float* xyz    = (const float*)d_in[0];
  const float* points = (const float*)d_in[1];
  const float* ts     = (const float*)d_in[2];
  const float* te_w1  = (const float*)d_in[3];
  const float* te_b1  = (const float*)d_in[4];
  const float* te_w2  = (const float*)d_in[5];
  const float* te_b2  = (const float*)d_in[6];
  const float* tf_w1  = (const float*)d_in[7];
  const float* tf_b1  = (const float*)d_in[8];
  const float* tf_w2  = (const float*)d_in[9];
  const float* tf_b2  = (const float*)d_in[10];
  const float* c1_w   = (const float*)d_in[11];
  const float* c1_b   = (const float*)d_in[12];
  const float* bn1_g  = (const float*)d_in[13];
  const float* bn1_b  = (const float*)d_in[14];
  const float* c2_w   = (const float*)d_in[15];
  const float* c2_b   = (const float*)d_in[16];
  const float* bn2_g  = (const float*)d_in[17];
  const float* bn2_b  = (const float*)d_in[18];
  const float* c3_w   = (const float*)d_in[19];
  const float* c3_b   = (const float*)d_in[20];
  const float* bn3_g  = (const float*)d_in[21];
  const float* bn3_b  = (const float*)d_in[22];
  const float* q_w    = (const float*)d_in[23];
  const float* q_b    = (const float*)d_in[24];
  const float* k_w    = (const float*)d_in[25];
  const float* k_b    = (const float*)d_in[26];
  const float* v_w    = (const float*)d_in[27];
  const float* v_b    = (const float*)d_in[28];
  const float* o_w    = (const float*)d_in[29];
  const float* o_b    = (const float*)d_in[30];
  const float* scalew = (const float*)d_in[31];
  const float* ln_g   = (const float*)d_in[32];
  const float* ln_b   = (const float*)d_in[33];

  float* outXyz = (float*)d_out;                     // (B,3,2048)
  float* outLN  = (float*)d_out + 8 * 3 * 2048;      // (B,256,2048)

  char* ws = (char*)d_ws;
  size_t off = 0;
  auto alloc = [&](size_t bytes) -> void* {
    void* p = ws + off;
    off = (off + bytes + 255) & ~(size_t)255;
    return p;
  };
  float*  stats   = (float*)alloc(16 * sizeof(float));
  int*    sampIdx = (int*)alloc(8 * 2048 * sizeof(int));
  int*    nidx    = (int*)alloc((size_t)8 * 2048 * 32 * sizeof(int));
  __bf16* comb    = (__bf16*)alloc((size_t)65536 * 96 * 2);
  __bf16* X1      = (__bf16*)alloc((size_t)65536 * 128 * 2);
  __bf16* ptsB    = (__bf16*)alloc((size_t)65536 * 64 * 2);
  __bf16* tfw1p   = (__bf16*)alloc(96 * 128 * 2);
  __bf16* tfw2b   = (__bf16*)alloc(128 * 64 * 2);
  __bf16* c1t     = (__bf16*)alloc(96 * 64 * 2);
  __bf16* c2t     = (__bf16*)alloc(64 * 128 * 2);
  __bf16* c3t     = (__bf16*)alloc(128 * 256 * 2);
  __bf16* qwb     = (__bf16*)alloc(256 * 256 * 2);
  __bf16* kwb     = (__bf16*)alloc(256 * 256 * 2);
  __bf16* vwb     = (__bf16*)alloc(256 * 256 * 2);
  __bf16* owb     = (__bf16*)alloc(256 * 256 * 2);
  float*  s1 = (float*)alloc(64 * 4);   float* t1 = (float*)alloc(64 * 4);
  float*  s2 = (float*)alloc(128 * 4);  float* t2 = (float*)alloc(128 * 4);
  float*  s3 = (float*)alloc(256 * 4);  float* t3 = (float*)alloc(256 * 4);
  __bf16* Xc   = (__bf16*)alloc((size_t)131072 * 96 * 2);
  __bf16* H1   = (__bf16*)alloc((size_t)131072 * 64 * 2);
  __bf16* H2   = (__bf16*)alloc((size_t)131072 * 128 * 2);
  __bf16* H3   = (__bf16*)alloc((size_t)131072 * 256 * 2);
  float*  feat = (float*)alloc((size_t)16384 * 256 * 4);
  __bf16* wx   = (__bf16*)alloc((size_t)4096 * 256 * 2);
  __bf16* Qb   = (__bf16*)alloc((size_t)4096 * 256 * 2);
  __bf16* Kb   = (__bf16*)alloc((size_t)4096 * 256 * 2);
  __bf16* Vb   = (__bf16*)alloc((size_t)4096 * 256 * 2);
  __bf16* Kt   = (__bf16*)alloc((size_t)8 * 4 * 64 * 512 * 2);
  float*  Smat = (float*)alloc((size_t)8 * 4 * 512 * 512 * 4);
  __bf16* Pmat = (__bf16*)alloc((size_t)8 * 4 * 512 * 512 * 2);
  __bf16* attc = (__bf16*)alloc((size_t)4096 * 256 * 2);
  float*  atto = (float*)alloc((size_t)4096 * 256 * 4);
  float*  msb  = (float*)alloc((size_t)8 * 2048 * 256 * 4);

  const float inv = 1.0f / sqrtf(1.0f + 1e-5f);

  // --- weight prep ---
  k_pad_bf16 <<<(96 * 128 + 255) / 256, 256, 0, stream>>>(tf_w1, tfw1p, 72, 96, 128);
  k_cast_bf16<<<(128 * 64 + 255) / 256, 256, 0, stream>>>(tf_w2, tfw2b, 128 * 64);
  k_tpad_bf16<<<(96 * 64 + 255) / 256, 256, 0, stream>>>(c1_w, c1t, 64, 67, 96);
  k_tpad_bf16<<<(64 * 128 + 255) / 256, 256, 0, stream>>>(c2_w, c2t, 128, 64, 64);
  k_tpad_bf16<<<(128 * 256 + 255) / 256, 256, 0, stream>>>(c3_w, c3t, 256, 128, 128);
  k_cast_bf16<<<256, 256, 0, stream>>>(q_w, qwb, 65536);
  k_cast_bf16<<<256, 256, 0, stream>>>(k_w, kwb, 65536);
  k_cast_bf16<<<256, 256, 0, stream>>>(v_w, vwb, 65536);
  k_cast_bf16<<<256, 256, 0, stream>>>(o_w, owb, 65536);
  k_bnaff<<<1, 256, 0, stream>>>(bn1_g, bn1_b, s1, t1, 64, inv);
  k_bnaff<<<1, 256, 0, stream>>>(bn2_g, bn2_b, s2, t2, 128, inv);
  k_bnaff<<<1, 256, 0, stream>>>(bn3_g, bn3_b, s3, t3, 256, inv);

  // --- time embed + transform MLP (WMMA GEMMs) ---
  k_comb<<<(65536 + 255) / 256, 256, 0, stream>>>(ts, points, te_w1, te_b1, te_w2, te_b2, comb);
  gemm(stream, comb, tfw1p, X1,  tf_b1, nullptr, nullptr, 65536, 128, 96, 96, 128, 128, 1, 1);
  gemm(stream, X1,   tfw2b, ptsB, tf_b2, nullptr, nullptr, 65536, 64, 128, 128, 64, 64, 0, 1);

  // --- sampling + KNN ---
  k_stats<<<8, 256, 0, stream>>>(ts, stats);
  k_sample<<<8, 512, 0, stream>>>(ts, xyz, stats, sampIdx, outXyz);
  k_knn<<<dim3(8, 8), 256, 0, stream>>>(xyz, outXyz, nidx);

  // --- grouped features as batched GEMMs over 4096-point chunks ---
  for (int ch = 0; ch < 4; ++ch) {
    int p0 = ch * 4096;
    k_gather<<<(4096 * 32 + 255) / 256, 256, 0, stream>>>(xyz, outXyz, ptsB, nidx, Xc, p0);
    gemm(stream, Xc, c1t, H1, c1_b, s1, t1, 131072, 64, 96, 96, 64, 64, 1, 1);
    gemm(stream, H1, c2t, H2, c2_b, s2, t2, 131072, 128, 64, 64, 128, 128, 1, 1);
    gemm(stream, H2, c3t, H3, c3_b, s3, t3, 131072, 256, 128, 128, 256, 256, 0, 1);
    k_maxpool<<<(4096 * 256 + 255) / 256, 256, 0, stream>>>(H3, feat, p0);
  }

  // --- multi-scale attention ---
  const int scales[3] = {4, 8, 16};
  for (int si = 0; si < 3; ++si) {
    int sc = scales[si];
    int nw = 2048 / sc;
    int Mq = 8 * nw;
    k_pool<<<(8 * nw * 256 + 255) / 256, 256, 0, stream>>>(feat, wx, nw, sc);
    gemm(stream, wx, qwb, Qb, q_b, nullptr, nullptr, Mq, 256, 256, 256, 256, 256, 0, 1);
    gemm(stream, wx, kwb, Kb, k_b, nullptr, nullptr, Mq, 256, 256, 256, 256, 256, 0, 1);
    gemm(stream, wx, vwb, Vb, v_b, nullptr, nullptr, Mq, 256, 256, 256, 256, 256, 0, 1);
    k_transK<<<(8 * 4 * 64 * nw + 255) / 256, 256, 0, stream>>>(Kb, Kt, nw);
    // scores = Q @ K^T, batched over (B, H)
    gemm(stream, Qb, Kt, Smat, nullptr, nullptr, nullptr, nw, nw, 64, 256, nw, nw, 0, 0,
         4, 64, (long long)64 * nw, (long long)nw * nw,
         8, (long long)nw * 256, (long long)4 * 64 * nw, (long long)4 * nw * nw);
    k_softmax<<<8 * 4 * nw, 256, 0, stream>>>(Smat, Pmat, nw);
    // att = P @ V, batched over (B, H)
    gemm(stream, Pmat, Vb, attc, nullptr, nullptr, nullptr, nw, 64, nw, nw, 256, 256, 0, 1,
         4, (long long)nw * nw, 64, 64,
         8, (long long)4 * nw * nw, (long long)nw * 256, (long long)nw * 256);
    gemm(stream, attc, owb, atto, o_b, nullptr, nullptr, Mq, 256, 256, 256, 256, 256, 0, 0);
    k_interp<<<(8 * 2048 * 256 + 255) / 256, 256, 0, stream>>>(atto, scalew, msb, nw, si, si == 0 ? 1 : 0);
  }

  // --- residual + layernorm -> (B, 256, 2048) ---
  k_layernorm<<<8 * 2048, 256, 0, stream>>>(feat, msb, ln_g, ln_b, outLN);
}
